// AttentionLayer_60026462928906
// MI455X (gfx1250) — matrix operand order repaired and checked by reading, not compiled, not run on hardware
//
#include <hip/hip_runtime.h>

#define NH    8
#define DH    64
#define CDIM  512
#define LDIM  2304
#define BDIM  4
#define MROWS (BDIM * LDIM)   // 9216
#define NQKV  (3 * CDIM)      // 1536

typedef __attribute__((ext_vector_type(16))) __bf16 bf16x16;
typedef __attribute__((ext_vector_type(8)))  __bf16 bf16x8;
typedef __attribute__((ext_vector_type(8)))  float  f32x8;

union FragU { bf16x16 v; bf16x8 h[2]; };

// A-fragment (16x32 bf16, MxK): lane<16 -> M=lane, K in {0..7, 16..23};
// lane>=16 -> M=lane-16, K in {8..15, 24..31}. Two 16B chunks per lane.
__device__ __forceinline__ bf16x16 load_frag_a(const __bf16* base, int ld, int row0, int k0) {
  const int lane = threadIdx.x & 31;
  const int r = lane & 15, hf = lane >> 4;
  const __bf16* p = base + (size_t)(row0 + r) * ld + (k0 + hf * 8);
  FragU f;
  f.h[0] = *(const bf16x8*)(p);
  f.h[1] = *(const bf16x8*)(p + 16);
  return f.v;
}

// B-fragment (32x16 bf16, KxN) given B^T stored row-major (rows = N, contiguous K):
// lane<16 -> N=lane, K in {0..7, 16..23}; lane>=16 -> N=lane-16, K in {8..15, 24..31}.
__device__ __forceinline__ bf16x16 load_frag_b(const __bf16* base, int ld, int col0, int k0) {
  const int lane = threadIdx.x & 31;
  const int n = lane & 15, hf = lane >> 4;
  const __bf16* p = base + (size_t)(col0 + n) * ld + (k0 + hf * 8);
  FragU f;
  f.h[0] = *(const bf16x8*)(p);
  f.h[1] = *(const bf16x8*)(p + 16);
  return f.v;
}

__device__ __forceinline__ f32x8 wmma_bf16(bf16x16 a, bf16x16 b, f32x8 c) {
  return __builtin_amdgcn_wmma_f32_16x16x32_bf16(false, a, false, b, (short)0, c, false, false);
}

// ---------------- conversion kernels ----------------

__global__ __launch_bounds__(256) void cvt_f32_bf16_kernel(const float* __restrict__ in,
                                                           __bf16* __restrict__ out, int n) {
  int i = blockIdx.x * blockDim.x + threadIdx.x;
  if (i < n) out[i] = (__bf16)in[i];
}

// out[c * rows + r] = in[r * cols + c]  (W [K,N] row-major -> W^T [N,K] bf16)
__global__ void transpose_cvt_kernel(const float* __restrict__ in, __bf16* __restrict__ out,
                                     int rows, int cols) {
  int c = blockIdx.x * blockDim.x + threadIdx.x;
  int r = blockIdx.y * blockDim.y + threadIdx.y;
  if (r < rows && c < cols)
    out[(size_t)c * rows + r] = (__bf16)in[(size_t)r * cols + c];
}

// ---------------- QKV projection GEMM ----------------
// 64x64 tile per wave: 16 WMMAs per 8 fragment loads (32 flop/L2-byte).
// qkv = xb[9216,512] * Wqkv + bqkv ; scatter Q,K [bh,L,64] (Q pre-scaled 1/8), V^T [bh,64,L]

__global__ __launch_bounds__(256)
void qkv_gemm_kernel(const __bf16* __restrict__ xb, const __bf16* __restrict__ wT,
                     const float* __restrict__ bias, __bf16* __restrict__ Qb,
                     __bf16* __restrict__ Kb, __bf16* __restrict__ Vt) {
  const int wid = blockIdx.x * 8 + (threadIdx.x >> 5);
  const int NT = NQKV / 64;                       // 24
  if (wid >= (MROWS / 64) * NT) return;
  const int m0 = (wid / NT) * 64;
  const int n0 = (wid % NT) * 64;
  f32x8 acc[4][4] = {};
  for (int k = 0; k < CDIM; k += 32) {
    bf16x16 af[4], bfr[4];
#pragma unroll
    for (int i = 0; i < 4; ++i) af[i] = load_frag_a(xb, CDIM, m0 + 16 * i, k);
#pragma unroll
    for (int j = 0; j < 4; ++j) bfr[j] = load_frag_b(wT, CDIM, n0 + 16 * j, k);
#pragma unroll
    for (int i = 0; i < 4; ++i)
#pragma unroll
      for (int j = 0; j < 4; ++j) acc[i][j] = wmma_bf16(af[i], bfr[j], acc[i][j]);
  }
  const int lane = threadIdx.x & 31, hf = lane >> 4, n15 = lane & 15;
  for (int i = 0; i < 4; ++i)
    for (int j = 0; j < 4; ++j)
      for (int r = 0; r < 8; ++r) {
        const int m = m0 + 16 * i + r + 8 * hf;
        const int n = n0 + 16 * j + n15;
        const float v = acc[i][j][r] + bias[n];
        const int bb = m / LDIM, l = m % LDIM;
        if (n < CDIM) {
          const int h = n >> 6, d = n & 63;
          Qb[(((size_t)(bb * NH + h)) * LDIM + l) * DH + d] = (__bf16)(v * 0.125f);
        } else if (n < 2 * CDIM) {
          const int nn = n - CDIM, h = nn >> 6, d = nn & 63;
          Kb[(((size_t)(bb * NH + h)) * LDIM + l) * DH + d] = (__bf16)v;
        } else {
          const int nn = n - 2 * CDIM, h = nn >> 6, d = nn & 63;
          Vt[(((size_t)(bb * NH + h)) * DH + d) * LDIM + l] = (__bf16)v;
        }
      }
}

// ---------------- flash attention ----------------
// One wave = 32 Q rows of one (b,h). Online softmax over 72 KV tiles of 32.
// 2304 waves keeps latency hiding; K/V are L2-resident.

__global__ __launch_bounds__(256)
void attn_kernel(const __bf16* __restrict__ Qb, const __bf16* __restrict__ Kb,
                 const __bf16* __restrict__ Vt, __bf16* __restrict__ ao) {
  __shared__ __attribute__((aligned(16))) __bf16 pbuf[8][32][40];
  const int wv = threadIdx.x >> 5;
  const int wid = blockIdx.x * 8 + wv;
  const int QT = LDIM / 32;                        // 72
  if (wid >= BDIM * NH * QT) return;
  const int bh = wid / QT, qt = wid % QT;
  const int b = bh / NH, h = bh % NH;
  const int m0 = qt * 32;
  const __bf16* Q = Qb + (size_t)bh * LDIM * DH;
  const __bf16* K = Kb + (size_t)bh * LDIM * DH;
  const __bf16* V = Vt + (size_t)bh * DH * LDIM;
  const int lane = threadIdx.x & 31, hf = lane >> 4, n15 = lane & 15;

  bf16x16 aq[2][2];
  for (int i = 0; i < 2; ++i)
    for (int s = 0; s < 2; ++s)
      aq[i][s] = load_frag_a(Q, DH, m0 + 16 * i, 32 * s);

  f32x8 o[2][4] = {};
  float mrun[2][8], lrun[2][8];
  for (int i = 0; i < 2; ++i)
    for (int r = 0; r < 8; ++r) { mrun[i][r] = -INFINITY; lrun[i][r] = 0.0f; }

  __bf16 (*pw)[40] = pbuf[wv];

  for (int kt = 0; kt < LDIM; kt += 32) {
    // S = Q * K^T  (32x32 tile, f32)
    f32x8 s[2][2] = {};
#pragma unroll
    for (int ss = 0; ss < 2; ++ss) {
      bf16x16 bk0 = load_frag_b(K, DH, kt, 32 * ss);
      bf16x16 bk1 = load_frag_b(K, DH, kt + 16, 32 * ss);
      s[0][0] = wmma_bf16(aq[0][ss], bk0, s[0][0]);
      s[0][1] = wmma_bf16(aq[0][ss], bk1, s[0][1]);
      s[1][0] = wmma_bf16(aq[1][ss], bk0, s[1][0]);
      s[1][1] = wmma_bf16(aq[1][ss], bk1, s[1][1]);
    }
    // online softmax (rows live along 16-lane halves; 16-lane xor reductions)
    for (int i = 0; i < 2; ++i) {
      for (int r = 0; r < 8; ++r) {
        const float v0 = s[i][0][r], v1 = s[i][1][r];
        float mx = fmaxf(v0, v1);
        for (int off = 8; off >= 1; off >>= 1) mx = fmaxf(mx, __shfl_xor(mx, off, 32));
        const float mnew = fmaxf(mrun[i][r], mx);
        const float alpha = __expf(mrun[i][r] - mnew);
        const float p0 = __expf(v0 - mnew);
        const float p1 = __expf(v1 - mnew);
        float rs = p0 + p1;
        for (int off = 8; off >= 1; off >>= 1) rs += __shfl_xor(rs, off, 32);
        lrun[i][r] = lrun[i][r] * alpha + rs;
        mrun[i][r] = mnew;
        for (int j = 0; j < 4; ++j) o[i][j][r] *= alpha;
        const int ml = 16 * i + r + 8 * hf;
        pw[ml][n15]      = (__bf16)p0;
        pw[ml][16 + n15] = (__bf16)p1;
      }
    }
    // wave-private LDS round-trip: C-layout P -> A-fragment layout
    asm volatile("s_wait_dscnt 0x0" ::: "memory");
    bf16x16 ap0 = load_frag_a(&pw[0][0], 40, 0, 0);
    bf16x16 ap1 = load_frag_a(&pw[0][0], 40, 16, 0);
#pragma unroll
    for (int j = 0; j < 4; ++j) {
      bf16x16 bv = load_frag_b(V, LDIM, 16 * j, kt);  // V^T rows = output dims, contiguous over kv
      o[0][j] = wmma_bf16(ap0, bv, o[0][j]);
      o[1][j] = wmma_bf16(ap1, bv, o[1][j]);
    }
  }

  // normalize and store attn output as [B*L, C] bf16 (input to out-projection)
  for (int i = 0; i < 2; ++i)
    for (int r = 0; r < 8; ++r) {
      const float inv = 1.0f / lrun[i][r];
      const int l_idx = m0 + 16 * i + r + 8 * hf;
      const size_t rowoff = ((size_t)(b * LDIM + l_idx)) * CDIM + h * DH;
      for (int j = 0; j < 4; ++j)
        ao[rowoff + 16 * j + n15] = (__bf16)(o[i][j][r] * inv);
    }
}

// ---------------- output projection + residual ----------------
// 64x32 tile per wave (8 WMMAs / 6 fragment loads), 2304 waves.

__global__ __launch_bounds__(256)
void out_proj_kernel(const __bf16* __restrict__ ab, const __bf16* __restrict__ woT,
                     const float* __restrict__ bo, const float* __restrict__ x,
                     float* __restrict__ out) {
  const int wid = blockIdx.x * 8 + (threadIdx.x >> 5);
  const int NT = CDIM / 32;                        // 16
  if (wid >= (MROWS / 64) * NT) return;
  const int m0 = (wid / NT) * 64;
  const int n0 = (wid % NT) * 32;
  f32x8 acc[4][2] = {};
  for (int k = 0; k < CDIM; k += 32) {
    bf16x16 af[4], bfr[2];
#pragma unroll
    for (int i = 0; i < 4; ++i) af[i] = load_frag_a(ab, CDIM, m0 + 16 * i, k);
#pragma unroll
    for (int j = 0; j < 2; ++j) bfr[j] = load_frag_b(woT, CDIM, n0 + 16 * j, k);
#pragma unroll
    for (int i = 0; i < 4; ++i)
#pragma unroll
      for (int j = 0; j < 2; ++j) acc[i][j] = wmma_bf16(af[i], bfr[j], acc[i][j]);
  }
  const int lane = threadIdx.x & 31, hf = lane >> 4, n15 = lane & 15;
  for (int i = 0; i < 4; ++i)
    for (int j = 0; j < 2; ++j)
      for (int r = 0; r < 8; ++r) {
        const int m = m0 + 16 * i + r + 8 * hf;
        const int n = n0 + 16 * j + n15;
        const size_t idx = (size_t)m * CDIM + n;
        out[idx] = acc[i][j][r] + bo[n] + x[idx];
      }
}

// ---------------- launch ----------------

extern "C" void kernel_launch(void* const* d_in, const int* in_sizes, int n_in,
                              void* d_out, int out_size, void* d_ws, size_t ws_size,
                              hipStream_t stream) {
  const float* x    = (const float*)d_in[0];
  const float* Wqkv = (const float*)d_in[1];
  const float* bqkv = (const float*)d_in[2];
  const float* Wo   = (const float*)d_in[3];
  const float* bo   = (const float*)d_in[4];
  float* out = (float*)d_out;

  char* ws = (char*)d_ws;
  const size_t SZ_XB  = (size_t)MROWS * CDIM * 2;          // 9,437,184
  const size_t SZ_WQT = (size_t)NQKV * CDIM * 2;           // 1,572,864
  const size_t SZ_WOT = (size_t)CDIM * CDIM * 2;           //   524,288
  const size_t SZ_QKV = (size_t)BDIM * NH * LDIM * DH * 2; // 9,437,184 each

  size_t off = 0;
  __bf16* xb  = (__bf16*)(ws + off); off += SZ_XB;
  __bf16* wqt = (__bf16*)(ws + off); off += SZ_WQT;
  __bf16* wot = (__bf16*)(ws + off); off += SZ_WOT;
  __bf16* Qb  = (__bf16*)(ws + off); off += SZ_QKV;
  __bf16* Kb  = (__bf16*)(ws + off); off += SZ_QKV;
  __bf16* Vt  = (__bf16*)(ws + off); off += SZ_QKV;
  __bf16* ao  = (__bf16*)(ws + off); off += SZ_XB;
  if (ws_size < off) return;

  const int nx = MROWS * CDIM;                                  // 4,718,592
  cvt_f32_bf16_kernel<<<nx / 256, 256, 0, stream>>>(x, xb, nx);
  transpose_cvt_kernel<<<dim3(NQKV / 16, CDIM / 16), dim3(16, 16), 0, stream>>>(Wqkv, wqt, CDIM, NQKV);
  transpose_cvt_kernel<<<dim3(CDIM / 16, CDIM / 16), dim3(16, 16), 0, stream>>>(Wo, wot, CDIM, CDIM);

  const int qkv_waves = (MROWS / 64) * (NQKV / 64);             // 3456
  qkv_gemm_kernel<<<qkv_waves / 8, 256, 0, stream>>>(xb, wqt, bqkv, Qb, Kb, Vt);

  const int attn_waves = BDIM * NH * (LDIM / 32);               // 2304
  attn_kernel<<<attn_waves / 8, 256, 0, stream>>>(Qb, Kb, Vt, ao);

  const int op_waves = (MROWS / 64) * (CDIM / 32);              // 2304
  out_proj_kernel<<<op_waves / 8, 256, 0, stream>>>(ao, wot, bo, x, out);
}
